// SeismicConvLSTM_22050362097738
// MI455X (gfx1250) — matrix operand order repaired and checked by reading, not compiled, not run on hardware
//
#include <hip/hip_runtime.h>
#include <hip/hip_bf16.h>

// ---------------- problem constants ----------------
#define B_    32
#define T_    20
#define INC_  3
#define W_    256
#define HID_  64
#define G_    256        // 4*HID gates
#define N_    8192       // B*W spatial columns per timestep
#define KP0   224        // im2col K layer0: 67*3=201 padded to 7*32
#define KP1   384        // im2col K layer1: 128*3 = 12*32
#define FC1K  327680     // T*HID*W
#define FC1N  512
#define OUTD  3
#define KSPLIT 64        // grid.y for FC1 split-K; 4 sub-splits per block
#define FC1_STEPS 320    // (FC1K/4) / (KSPLIT*4)

// ---------------- device helpers ----------------
__device__ __forceinline__ unsigned short f2bf(float f) {
  unsigned int u = __float_as_uint(f);
  u += 0x7FFFu + ((u >> 16) & 1u);          // round-to-nearest-even
  return (unsigned short)(u >> 16);
}
__device__ __forceinline__ float sigf(float x) { return 1.0f / (1.0f + expf(-x)); }

// ---------------- elementwise kernels ----------------
__global__ void k_zero(float* p, int n) {
  int i = blockIdx.x * blockDim.x + threadIdx.x;
  if (i < n) p[i] = 0.0f;
}

// conv weight (G, cin, 3, 3) fp32 -> bf16 packed (G, kp) with k = c*3+kw, kh=1 row only
__global__ void k_pack_w(const float* __restrict__ w, unsigned short* __restrict__ wp,
                         int cin, int kp) {
  int i = blockIdx.x * blockDim.x + threadIdx.x;
  if (i >= G_ * kp) return;
  int g = i / kp, k = i - g * kp;
  float v = 0.0f;
  if (k < cin * 3) {
    int c = k / 3, kw = k - 3 * c;
    v = w[((g * cin + c) * 3 + 1) * 3 + kw];
  }
  wp[i] = f2bf(v);
}

// build column-major im2col (bf16): zc[n*kp + k], n = b*W + w, k = c*3 + kw
// channels = concat(xt (cinx ch), h (HID ch)); SAME padding over W
__global__ void k_zcol(const float* __restrict__ xt, int xbs, int cinx,
                       const float* __restrict__ h, unsigned short* __restrict__ zc, int kp) {
  int i = blockIdx.x * blockDim.x + threadIdx.x;
  if (i >= N_ * kp) return;
  int nn = i / kp, k = i - nn * kp;
  int b = nn >> 8, wp_ = nn & 255;
  int c = k / 3, kw = k - 3 * c;
  int ws = wp_ + kw - 1;
  float v = 0.0f;
  if ((unsigned)ws < (unsigned)W_) {
    if (c < cinx)            v = xt[b * xbs + c * W_ + ws];
    else if (c < cinx + HID_) v = h[(b * HID_ + (c - cinx)) * W_ + ws];
  }
  zc[i] = f2bf(v);
}

// ---------------- conv-gates GEMM: gates(256 x 8192) = Wp(256 x kp) * Zc^T ----------------
// 8 waves/block, each wave owns one 16x16 tile; grid (N_/16, G_/128)
__global__ void __launch_bounds__(256)
k_gates_gemm(const unsigned short* __restrict__ wp, const unsigned short* __restrict__ zc,
             float* __restrict__ gates, int kp, int ksteps) {
#if defined(__HIP_DEVICE_COMPILE__) && __has_builtin(__builtin_amdgcn_wmma_f32_16x16x32_bf16)
  typedef __bf16 bf16t;
  typedef __attribute__((ext_vector_type(16))) bf16t v16bf;
  typedef __attribute__((ext_vector_type(8)))  unsigned short v8us;
  typedef __attribute__((ext_vector_type(16))) unsigned short v16us;
  typedef __attribute__((ext_vector_type(8)))  float v8f;

  int lane  = threadIdx.x & 31;
  int wv    = threadIdx.x >> 5;
  int mtile = blockIdx.y * 8 + wv;     // 0..15
  int ntile = blockIdx.x;              // 0..511
  int l15 = lane & 15;
  int hi  = lane >> 4;

  // A (16x32 bf16): lane row = mtile*16 + l15; 16 consecutive K at k + 16*hi
  const unsigned short* ap = wp + (mtile * 16 + l15) * kp + hi * 16;
  // B (32x16 bf16): lane col = ntile*16 + l15; 16 consecutive K at k + 16*hi
  const unsigned short* bp = zc + (ntile * 16 + l15) * kp + hi * 16;

  v8f acc = {};
  for (int s = 0; s < ksteps; ++s) {
    v16us au = *(const v16us*)(ap);
    v16us bu = *(const v16us*)(bp);
    v16bf a = __builtin_bit_cast(v16bf, au);
    v16bf b = __builtin_bit_cast(v16bf, bu);
    acc = __builtin_amdgcn_wmma_f32_16x16x32_bf16(false, a, false, b, (short)0, acc,
                                                  false, false);
    ap += 32; bp += 32;
  }
  // C layout: VGPR r -> row = r + 8*hi; col = l15
  int g0 = mtile * 16 + hi * 8;
  int n  = ntile * 16 + l15;
#pragma unroll
  for (int r = 0; r < 8; ++r) gates[(g0 + r) * N_ + n] = acc[r];
#endif
}

// ---------------- LSTM pointwise ----------------
__global__ void k_lstm(const float* __restrict__ gates, const float* __restrict__ bias,
                       float* __restrict__ c, float* __restrict__ h,
                       float* __restrict__ hseq_t) {
  int i = blockIdx.x * blockDim.x + threadIdx.x;
  if (i >= B_ * HID_ * W_) return;
  int b   = i / (HID_ * W_);
  int rem = i - b * (HID_ * W_);
  int hc  = rem / W_;
  int w   = rem - hc * W_;
  int n   = b * W_ + w;
  float gi = gates[(0 * HID_ + hc) * N_ + n] + bias[0 * HID_ + hc];
  float gf = gates[(1 * HID_ + hc) * N_ + n] + bias[1 * HID_ + hc];
  float go = gates[(2 * HID_ + hc) * N_ + n] + bias[2 * HID_ + hc];
  float gg = gates[(3 * HID_ + hc) * N_ + n] + bias[3 * HID_ + hc];
  float cn = sigf(gf) * c[i] + sigf(gi) * tanhf(gg);
  float hn = sigf(go) * tanhf(cn);
  c[i] = cn;
  h[i] = hn;
  hseq_t[b * (T_ * HID_ * W_) + hc * W_ + w] = hn;
}

// ---------------- FC1 (fp32, split-K, f32 WMMA) ----------------
__global__ void k_initacc(float* acc, const float* b1) {
  int i = blockIdx.x * blockDim.x + threadIdx.x;
  if (i < B_ * FC1N) acc[i] = b1[i & (FC1N - 1)];
}

// out(32 x 512) += flat(32 x FC1K) * w(512 x FC1K)^T   via V_WMMA_F32_16X16X4_F32
// grid (FC1N/16, KSPLIT), block 256: 8 waves = 2 M-tiles x 4 K-sub-splits
__global__ void __launch_bounds__(256)
k_fc1(const float* __restrict__ flat, const float* __restrict__ w, float* __restrict__ acc) {
#if defined(__HIP_DEVICE_COMPILE__) && __has_builtin(__builtin_amdgcn_wmma_f32_16x16x4_f32)
  typedef __attribute__((ext_vector_type(2))) float v2f;
  typedef __attribute__((ext_vector_type(8))) float v8f;

  int lane  = threadIdx.x & 31;
  int wv    = threadIdx.x >> 5;
  int mtile = wv & 1;           // 2 M-tiles cover B=32
  int ksub  = wv >> 1;          // 4 K sub-splits per block
  int chunk = blockIdx.y * 4 + ksub;        // 0..255
  int k0    = chunk * (FC1_STEPS * 4);
  int l15 = lane & 15;
  int hi  = lane >> 4;

  // 32-bit A layout 16x4: lanes 0-15 K=0,1 ; lanes 16-31 K=2,3
  const float* ap = flat + (size_t)(mtile * 16 + l15) * FC1K + k0 + hi * 2;
  const float* bp = w    + (size_t)(blockIdx.x * 16 + l15) * FC1K + k0 + hi * 2;

  v8f c = {};
  for (int s = 0; s < FC1_STEPS; ++s) {
    __builtin_prefetch(bp + 256, 0, 0);     // global_prefetch_b8 down the weight stream
    v2f a = *(const v2f*)ap;
    v2f b = *(const v2f*)bp;
    c = __builtin_amdgcn_wmma_f32_16x16x4_f32(false, a, false, b, (short)0, c,
                                              false, false);
    ap += 4; bp += 4;
  }
  int b0 = mtile * 16 + hi * 8;
  int j  = blockIdx.x * 16 + l15;
#pragma unroll
  for (int r = 0; r < 8; ++r) atomicAdd(&acc[(b0 + r) * FC1N + j], c[r]);
#endif
}

// ---------------- head: relu -> fc2 -> sigmoid(ch2) ----------------
__global__ void k_head(const float* __restrict__ acc, const float* __restrict__ w2,
                       const float* __restrict__ b2, float* __restrict__ out) {
  int i = threadIdx.x;
  if (i >= B_ * OUTD) return;
  int b = i / OUTD, o = i - b * OUTD;
  float s = b2[o];
  for (int j = 0; j < FC1N; ++j)
    s += fmaxf(acc[b * FC1N + j], 0.0f) * w2[o * FC1N + j];
  if (o == 2) s = sigf(s);
  out[i] = s;
}

// ---------------- host orchestration ----------------
extern "C" void kernel_launch(void* const* d_in, const int* in_sizes, int n_in,
                              void* d_out, int out_size, void* d_ws, size_t ws_size,
                              hipStream_t stream) {
  (void)in_sizes; (void)n_in; (void)out_size; (void)ws_size;
  const float* x    = (const float*)d_in[0];
  const float* cw0  = (const float*)d_in[1];
  const float* cb0  = (const float*)d_in[2];
  const float* cw1  = (const float*)d_in[3];
  const float* cb1  = (const float*)d_in[4];
  const float* fc1w = (const float*)d_in[5];
  const float* fc1b = (const float*)d_in[6];
  const float* fc2w = (const float*)d_in[7];
  const float* fc2b = (const float*)d_in[8];

  size_t off = 0;
  char* base = (char*)d_ws;
  auto take = [&](size_t bytes) -> char* {
    char* p = base + off;
    off += (bytes + 255) & ~(size_t)255;
    return p;
  };
  const int HCW = B_ * HID_ * W_;          // 524288 (2 MB fp32, 256B multiple)
  float* h0 = (float*)take((size_t)HCW * 4);
  float* c0 = (float*)take((size_t)HCW * 4);
  float* h1 = (float*)take((size_t)HCW * 4);
  float* c1 = (float*)take((size_t)HCW * 4);
  float* h0seq = (float*)take((size_t)B_ * T_ * HID_ * W_ * 4);
  float* h1seq = (float*)take((size_t)B_ * T_ * HID_ * W_ * 4);
  float* gates = (float*)take((size_t)G_ * N_ * 4);
  float* acc   = (float*)take((size_t)B_ * FC1N * 4);
  unsigned short* wp0 = (unsigned short*)take((size_t)G_ * KP0 * 2);
  unsigned short* wp1 = (unsigned short*)take((size_t)G_ * KP1 * 2);
  unsigned short* zc  = (unsigned short*)take((size_t)N_ * KP1 * 2);

  // zero h0/c0/h1/c1 (contiguous: each region is a 256B multiple)
  k_zero<<<(4 * HCW + 255) / 256, 256, 0, stream>>>(h0, 4 * HCW);

  // pack conv weights to bf16 im2col layout
  k_pack_w<<<(G_ * KP0 + 255) / 256, 256, 0, stream>>>(cw0, wp0, INC_ + HID_, KP0);
  k_pack_w<<<(G_ * KP1 + 255) / 256, 256, 0, stream>>>(cw1, wp1, 2 * HID_, KP1);

  dim3 gemm_grid(N_ / 16, G_ / 128);   // (512, 2)

  // layer 0
  for (int t = 0; t < T_; ++t) {
    k_zcol<<<(N_ * KP0 + 255) / 256, 256, 0, stream>>>(
        x + (size_t)t * INC_ * W_, T_ * INC_ * W_, INC_, h0, zc, KP0);
    k_gates_gemm<<<gemm_grid, 256, 0, stream>>>(wp0, zc, gates, KP0, KP0 / 32);
    k_lstm<<<(HCW + 255) / 256, 256, 0, stream>>>(gates, cb0, c0, h0,
                                                  h0seq + (size_t)t * HID_ * W_);
  }
  // layer 1
  for (int t = 0; t < T_; ++t) {
    k_zcol<<<(N_ * KP1 + 255) / 256, 256, 0, stream>>>(
        h0seq + (size_t)t * HID_ * W_, T_ * HID_ * W_, HID_, h1, zc, KP1);
    k_gates_gemm<<<gemm_grid, 256, 0, stream>>>(wp1, zc, gates, KP1, KP1 / 32);
    k_lstm<<<(HCW + 255) / 256, 256, 0, stream>>>(gates, cb1, c1, h1,
                                                  h1seq + (size_t)t * HID_ * W_);
  }

  // FC1 split-K (flat view of h1seq is exactly (B, T*HID*W) contiguous)
  k_initacc<<<(B_ * FC1N + 255) / 256, 256, 0, stream>>>(acc, fc1b);
  dim3 fc1_grid(FC1N / 16, KSPLIT);    // (32, 64)
  k_fc1<<<fc1_grid, 256, 0, stream>>>(h1seq, fc1w, acc);

  // head
  k_head<<<1, 128, 0, stream>>>(acc, fc2w, fc2b, (float*)d_out);
}
